// GlobalBottleneckAttention3D_42485816492662
// MI455X (gfx1250) — matrix operand order repaired and checked by reading, not compiled, not run on hardware
//
#include <hip/hip_runtime.h>
#include <math.h>

// ---------------- problem constants ----------------
#define Bc   8
#define Cc   512
#define Tt   16
#define Sc   1024          // T*H*W
#define NHc  8
#define HDc  64
#define HIDc 1024
#define Mrows (Bc*Sc)      // 8192
#define EPSc 1e-5f

// ---------------- types ----------------
typedef __attribute__((ext_vector_type(16))) __bf16 v16bf;
typedef __attribute__((ext_vector_type(8)))  float  v8f;
typedef __attribute__((ext_vector_type(4)))  unsigned int v4u;
typedef __attribute__((ext_vector_type(8)))  int    v8i;
typedef __attribute__((ext_vector_type(4)))  int    v4i;

static __device__ __forceinline__ unsigned short f2bits(float f) {
  union { float f; unsigned u; } v; v.f = f;
  unsigned u = v.u;
  unsigned r = u + 0x7FFFu + ((u >> 16) & 1u);   // round-to-nearest-even
  return (unsigned short)(r >> 16);
}
static __device__ __forceinline__ __bf16 bfbits(unsigned short h) {
  __bf16 b; __builtin_memcpy(&b, &h, 2); return b;
}
static __device__ __forceinline__ v8f wmma_bf16(v16bf a, v16bf b, v8f c) {
  return __builtin_amdgcn_wmma_f32_16x16x32_bf16(false, a, false, b, (short)0, c, false, false);
}

// ---------------- Tensor Data Mover (gfx1250) ----------------
#if defined(__gfx1250__) && __has_builtin(__builtin_amdgcn_tensor_load_to_lds)
#define HAVE_TDM 1
// 2D tile load: tile_d1 rows of tile_d0 bf16 elements, row stride stride0 (elems),
// from gptr (tile start) into contiguous LDS at lds_off.
static __device__ __forceinline__ void tdm_load_2d(unsigned lds_off, const void* gptr,
                                                   unsigned tensor_d0, unsigned tensor_d1,
                                                   unsigned tile_d0, unsigned tile_d1,
                                                   unsigned long long stride0) {
  unsigned long long ga = (unsigned long long)(uintptr_t)gptr;
  v4u g0;
  g0[0] = 1u;                                            // count=1, user-mode, no gather
  g0[1] = lds_off;                                       // LDS byte address
  g0[2] = (unsigned)ga;                                  // global_addr[31:0]
  g0[3] = (unsigned)((ga >> 32) & 0x1FFFFFFu) | (2u << 30); // global_addr[56:32], type=2
  v8i g1;
  g1[0] = (int)(1u << 16);                               // wg_mask=0, data_size=1 (2B)
  g1[1] = (int)((tensor_d0 & 0xFFFFu) << 16);            // tensor_dim0[15:0]
  g1[2] = (int)((tensor_d0 >> 16) | ((tensor_d1 & 0xFFFFu) << 16));
  g1[3] = (int)((tensor_d1 >> 16) | ((tile_d0 & 0xFFFFu) << 16));
  g1[4] = (int)(tile_d1 & 0xFFFFu);                      // tile_dim1, tile_dim2=0
  g1[5] = (int)(unsigned)(stride0 & 0xFFFFFFFFull);      // tensor_dim0_stride[31:0]
  g1[6] = (int)(unsigned)((stride0 >> 32) & 0xFFFFull);  // stride0[47:32], stride1 lo=0
  g1[7] = 0;                                             // stride1 hi = 0
  v4i z4 = {0, 0, 0, 0};
  v8i z8 = {0, 0, 0, 0, 0, 0, 0, 0};
  __builtin_amdgcn_tensor_load_to_lds(g0, g1, z4, z4, z8, 0);  // 6-arg (clang-23) form
}
static __device__ __forceinline__ unsigned lds_addr_of(const void* p) {
  return (unsigned)(uintptr_t)p;   // low 32 bits of generic LDS address = LDS offset
}
#endif

// ---------------- elementwise fp32 -> bf16 ----------------
__global__ __launch_bounds__(256) void k_f32_to_bf16(const float* __restrict__ in,
                                                     unsigned short* __restrict__ out, int n) {
  int i = blockIdx.x * 256 + threadIdx.x;
  if (i < n) out[i] = f2bits(in[i]);
}

// ---------------- LN1: x[B,C,S] -> tok fp32 [B,S,C] + xln bf16 [B,S,C] ----------------
__global__ __launch_bounds__(256) void k_ln1(const float* __restrict__ x,
                                             const float* __restrict__ w,
                                             const float* __restrict__ bias,
                                             float* __restrict__ tok,
                                             unsigned short* __restrict__ xln) {
  __shared__ float red[256];
  int bs = blockIdx.x;
  int b  = bs >> 10;
  int s  = bs & (Sc - 1);
  const float* xb = x + (size_t)b * Cc * Sc + s;
  int c0 = threadIdx.x, c1 = threadIdx.x + 256;
  float v0 = xb[(size_t)c0 * Sc];
  float v1 = xb[(size_t)c1 * Sc];

  red[threadIdx.x] = v0 + v1; __syncthreads();
  for (int off = 128; off > 0; off >>= 1) {
    if (threadIdx.x < off) red[threadIdx.x] += red[threadIdx.x + off];
    __syncthreads();
  }
  float mean = red[0] * (1.0f / Cc);
  __syncthreads();
  float d0 = v0 - mean, d1 = v1 - mean;
  red[threadIdx.x] = d0 * d0 + d1 * d1; __syncthreads();
  for (int off = 128; off > 0; off >>= 1) {
    if (threadIdx.x < off) red[threadIdx.x] += red[threadIdx.x + off];
    __syncthreads();
  }
  float rstd = rsqrtf(red[0] * (1.0f / Cc) + EPSc);

  size_t base = (size_t)bs * Cc;
  tok[base + c0] = v0;
  tok[base + c1] = v1;
  xln[base + c0] = f2bits(d0 * rstd * w[c0] + bias[c0]);
  xln[base + c1] = f2bits(d1 * rstd * w[c1] + bias[c1]);
}

// ---------------- LN2: tok fp32 [B,S,C] -> bf16 [B,S,C] ----------------
__global__ __launch_bounds__(256) void k_ln2(const float* __restrict__ tok,
                                             const float* __restrict__ w,
                                             const float* __restrict__ bias,
                                             unsigned short* __restrict__ out) {
  __shared__ float red[256];
  size_t base = (size_t)blockIdx.x * Cc;
  int c0 = threadIdx.x, c1 = threadIdx.x + 256;
  float v0 = tok[base + c0];
  float v1 = tok[base + c1];
  red[threadIdx.x] = v0 + v1; __syncthreads();
  for (int off = 128; off > 0; off >>= 1) {
    if (threadIdx.x < off) red[threadIdx.x] += red[threadIdx.x + off];
    __syncthreads();
  }
  float mean = red[0] * (1.0f / Cc);
  __syncthreads();
  float d0 = v0 - mean, d1 = v1 - mean;
  red[threadIdx.x] = d0 * d0 + d1 * d1; __syncthreads();
  for (int off = 128; off > 0; off >>= 1) {
    if (threadIdx.x < off) red[threadIdx.x] += red[threadIdx.x + off];
    __syncthreads();
  }
  float rstd = rsqrtf(red[0] * (1.0f / Cc) + EPSc);
  out[base + c0] = f2bits(d0 * rstd * w[c0] + bias[c0]);
  out[base + c1] = f2bits(d1 * rstd * w[c1] + bias[c1]);
}

// ---------------- tiled WMMA GEMM: C[M,N] = A[M,K](bf16) * W[N,K]^T(bf16) + bias ----
// mode 0: out bf16 ; mode 1: SiLU -> out bf16 ; mode 2: fp32 residual += (Res[M,N])
// TDM double-buffered: wave0 DMAs tile k+1 while all waves run WMMA on tile k.
#define BM 64
#define BN 64
#define BK 32
__global__ __launch_bounds__(128) void k_gemm_bf16(const unsigned short* __restrict__ A,
                                                   const unsigned short* __restrict__ W,
                                                   const float* __restrict__ bias,
                                                   unsigned short* __restrict__ OutBf,
                                                   float* __restrict__ Res,
                                                   int M, int N, int K, int mode) {
  __shared__ __attribute__((aligned(16))) unsigned short sA[2][BM][BK];
  __shared__ __attribute__((aligned(16))) unsigned short sB[2][BN][BK];

  int tid  = threadIdx.x;
  int wave = tid >> 5;
  int lane = tid & 31;
  int half = lane >> 4;
  int nl   = lane & 15;
  int bm = blockIdx.y * BM;
  int bn = blockIdx.x * BN;

  v8f acc[4];
  for (int i = 0; i < 4; ++i) acc[i] = (v8f){};

  int nk = K / BK;

#ifdef HAVE_TDM
  if (wave == 0) {
    tdm_load_2d(lds_addr_of(&sA[0][0][0]), A + (size_t)bm * K, K, M, BK, BM, K);
    tdm_load_2d(lds_addr_of(&sB[0][0][0]), W + (size_t)bn * K, K, N, BK, BN, K);
  }
#endif

  for (int kt = 0; kt < nk; ++kt) {
    int cur = kt & 1;
#ifdef HAVE_TDM
    if (wave == 0) {
      if (kt + 1 < nk) {
        int k1 = (kt + 1) * BK;
        tdm_load_2d(lds_addr_of(&sA[cur ^ 1][0][0]), A + (size_t)bm * K + k1, K, M, BK, BM, K);
        tdm_load_2d(lds_addr_of(&sB[cur ^ 1][0][0]), W + (size_t)bn * K + k1, K, N, BK, BN, K);
        __builtin_amdgcn_s_wait_tensorcnt((short)2);   // oldest pair (current tile) done
      } else {
        __builtin_amdgcn_s_wait_tensorcnt((short)0);
      }
    }
    __syncthreads();
#else
    int k0 = kt * BK;
    for (int it = 0; it < 4; ++it) {
      int idx = tid + it * 128;
      int r = idx >> 3;
      int c = (idx & 7) * 4;
      *(unsigned long long*)&sA[cur][r][c] =
          *(const unsigned long long*)(A + (size_t)(bm + r) * K + k0 + c);
      *(unsigned long long*)&sB[cur][r][c] =
          *(const unsigned long long*)(W + (size_t)(bn + r) * K + k0 + c);
    }
    __syncthreads();
#endif

    // A fragment: 16x32 half-wave K split (half0 -> K 0..15; half1 -> K 16..31)
    v16bf af;
    int am = wave * 16 + nl;
    int kb = half * 16;
    for (int i = 0; i < 8; ++i) {
      af[i]     = bfbits(sA[cur][am][kb + i]);
      af[8 + i] = bfbits(sA[cur][am][kb + 8 + i]);
    }
    for (int nt = 0; nt < 4; ++nt) {
      // B fragment: B[k][n] = W[n][k]; lanes 0-15 hold K 0..15, lanes 16-31 hold K 16..31
      v16bf bfr;
      int ncol = nt * 16 + nl;
      int kbb  = half * 16;
      for (int i = 0; i < 16; ++i) bfr[i] = bfbits(sB[cur][ncol][kbb + i]);
      acc[nt] = wmma_bf16(af, bfr, acc[nt]);
    }
    __syncthreads();   // all waves done reading buf[cur] before TDM refills it
  }

  // epilogue: C layout m = r + 8*half, n = lane%16 (per 16x16 tile)
  for (int nt = 0; nt < 4; ++nt) {
    for (int r = 0; r < 8; ++r) {
      int gm = bm + wave * 16 + r + 8 * half;
      int gn = bn + nt * 16 + nl;
      float v = acc[nt][r] + bias[gn];
      if (mode == 0) {
        OutBf[(size_t)gm * N + gn] = f2bits(v);
      } else if (mode == 1) {
        float sv = v / (1.0f + __expf(-v));   // SiLU
        OutBf[(size_t)gm * N + gn] = f2bits(sv);
      } else {
        Res[(size_t)gm * N + gn] += v;        // fp32 residual add
      }
    }
  }
}

// ---------------- flash-style attention, block-causal over 16 time blocks -----------
// qkv bf16 [B, S, 3C]; out bf16 [B, S, C]. One workgroup = (b, head, query time block).
__global__ __launch_bounds__(128) void k_attn(const unsigned short* __restrict__ qkv,
                                              unsigned short* __restrict__ attn_out) {
  __shared__ __attribute__((aligned(16))) unsigned short sQ[64][64];
  __shared__ __attribute__((aligned(16))) unsigned short sK[64][64];
  __shared__ __attribute__((aligned(16))) unsigned short sV[64][64];
  __shared__ __attribute__((aligned(16))) unsigned short sP[64][64];
  __shared__ float sS[64][64];
  __shared__ float sMax[64], sSum[64], sCorr[64];

  int bid = blockIdx.x;
  int tq = bid & (Tt - 1);
  int h  = (bid >> 4) & (NHc - 1);
  int b  = bid >> 7;
  int qbase = tq * 64;

  int tid  = threadIdx.x;
  int wave = tid >> 5;
  int lane = tid & 31;
  int half = lane >> 4;
  int nl   = lane & 15;

  // load Q block (64 x 64 bf16), row stride 3C in global
#ifdef HAVE_TDM
  if (wave == 0) {
    tdm_load_2d(lds_addr_of(&sQ[0][0]),
                qkv + ((size_t)b * Sc + qbase) * (3 * Cc) + h * HDc,
                3 * Cc, Sc, 64, 64, 3 * Cc);
    __builtin_amdgcn_s_wait_tensorcnt((short)0);
  }
#else
  for (int i = tid; i < 1024; i += 128) {
    int r = i >> 4, c4 = (i & 15) * 4;
    *(unsigned long long*)&sQ[r][c4] =
        *(const unsigned long long*)(qkv + ((size_t)b * Sc + qbase + r) * (3 * Cc) + h * HDc + c4);
  }
#endif
  if (tid < 64) { sMax[tid] = -1e30f; sSum[tid] = 0.0f; }
  __syncthreads();

  // Q fragments (two K-steps of 32 over head dim 64), loaded once
  v16bf qf[2];
  {
    int qm = wave * 16 + nl;
    int kb = half * 16;
    for (int ks = 0; ks < 2; ++ks)
      for (int i = 0; i < 8; ++i) {
        qf[ks][i]     = bfbits(sQ[qm][ks * 32 + kb + i]);
        qf[ks][8 + i] = bfbits(sQ[qm][ks * 32 + kb + 8 + i]);
      }
  }

  v8f o[4];
  for (int i = 0; i < 4; ++i) o[i] = (v8f){};

  for (int tk = 0; tk <= tq; ++tk) {   // block-causal: only key blocks with tk <= tq
    int kbase = tk * 64;
    __syncthreads();                   // previous iteration done with sK/sV/sP
#ifdef HAVE_TDM
    if (wave == 0) {
      const unsigned short* krow = qkv + ((size_t)b * Sc + kbase) * (3 * Cc) + h * HDc;
      tdm_load_2d(lds_addr_of(&sK[0][0]), krow + Cc,     3 * Cc, Sc, 64, 64, 3 * Cc);
      tdm_load_2d(lds_addr_of(&sV[0][0]), krow + 2 * Cc, 3 * Cc, Sc, 64, 64, 3 * Cc);
      __builtin_amdgcn_s_wait_tensorcnt((short)0);
    }
#else
    for (int i = tid; i < 1024; i += 128) {
      int r = i >> 4, c4 = (i & 15) * 4;
      size_t rowoff = ((size_t)b * Sc + kbase + r) * (3 * Cc);
      *(unsigned long long*)&sK[r][c4] =
          *(const unsigned long long*)(qkv + rowoff + Cc + h * HDc + c4);
      *(unsigned long long*)&sV[r][c4] =
          *(const unsigned long long*)(qkv + rowoff + 2 * Cc + h * HDc + c4);
    }
#endif
    __syncthreads();

    // scores: S = Q * K^T  (B[d][n] = K[n][d])
    v8f sc[4];
    for (int i = 0; i < 4; ++i) sc[i] = (v8f){};
    for (int ks = 0; ks < 2; ++ks) {
      for (int nt = 0; nt < 4; ++nt) {
        v16bf bfr;
        int ncol = nt * 16 + nl;
        int dkb  = ks * 32 + half * 16;
        for (int i = 0; i < 16; ++i) bfr[i] = bfbits(sK[ncol][dkb + i]);
        sc[nt] = wmma_bf16(qf[ks], bfr, sc[nt]);
      }
    }
    for (int nt = 0; nt < 4; ++nt)
      for (int r = 0; r < 8; ++r)
        sS[wave * 16 + r + 8 * half][nt * 16 + nl] = sc[nt][r] * 0.125f;  // 1/sqrt(64)
    __syncthreads();

    // online softmax row update
    if (tid < 64) {
      float mold = sMax[tid];
      float rmax = mold;
      for (int j = 0; j < 64; ++j) rmax = fmaxf(rmax, sS[tid][j]);
      float corr = __expf(mold - rmax);
      float rsum = 0.0f;
      for (int j = 0; j < 64; ++j) {
        float p = __expf(sS[tid][j] - rmax);
        rsum += p;
        sP[tid][j] = f2bits(p);
      }
      sMax[tid] = rmax;
      sSum[tid] = sSum[tid] * corr + rsum;
      sCorr[tid] = corr;
    }
    __syncthreads();

    // rescale accumulators, then O += P * V
    float cf[8];
    for (int r = 0; r < 8; ++r) cf[r] = sCorr[wave * 16 + r + 8 * half];
    for (int nt = 0; nt < 4; ++nt)
      for (int r = 0; r < 8; ++r) o[nt][r] *= cf[r];

    for (int ks = 0; ks < 2; ++ks) {
      v16bf pf;
      int pm = wave * 16 + nl;
      int kb = half * 16;
      for (int i = 0; i < 8; ++i) {
        pf[i]     = bfbits(sP[pm][ks * 32 + kb + i]);
        pf[8 + i] = bfbits(sP[pm][ks * 32 + kb + 8 + i]);
      }
      for (int nt = 0; nt < 4; ++nt) {
        v16bf vfr;
        int vn  = nt * 16 + nl;
        int vkb = ks * 32 + half * 16;
        for (int i = 0; i < 16; ++i) vfr[i] = bfbits(sV[vkb + i][vn]);
        o[nt] = wmma_bf16(pf, vfr, o[nt]);
      }
    }
  }

  __syncthreads();
  for (int nt = 0; nt < 4; ++nt)
    for (int r = 0; r < 8; ++r) {
      int row = wave * 16 + r + 8 * half;
      float invl = 1.0f / sSum[row];
      int d = nt * 16 + nl;
      attn_out[((size_t)b * Sc + qbase + row) * Cc + h * HDc + d] = f2bits(o[nt][r] * invl);
    }
}

// ---------------- final transpose: tok [B,S,C] fp32 -> out [B,C,S] fp32 -------------
__global__ __launch_bounds__(256) void k_out_transpose(const float* __restrict__ tok,
                                                       float* __restrict__ out) {
  size_t i = (size_t)blockIdx.x * 256 + threadIdx.x;   // output-linear index
  if (i >= (size_t)Bc * Cc * Sc) return;
  int s  = (int)(i & (Sc - 1));
  int c  = (int)((i >> 10) & (Cc - 1));
  int b  = (int)(i >> 19);
  out[i] = tok[((size_t)b * Sc + s) * Cc + c];
}

// ---------------- host-side launch ----------------
extern "C" void kernel_launch(void* const* d_in, const int* in_sizes, int n_in,
                              void* d_out, int out_size, void* d_ws, size_t ws_size,
                              hipStream_t stream) {
  const float* x         = (const float*)d_in[0];
  const float* ln1_w     = (const float*)d_in[1];
  const float* ln1_b     = (const float*)d_in[2];
  const float* in_proj_w = (const float*)d_in[3];
  const float* in_proj_b = (const float*)d_in[4];
  const float* out_w     = (const float*)d_in[5];
  const float* out_b     = (const float*)d_in[6];
  const float* ln2_w     = (const float*)d_in[7];
  const float* ln2_b     = (const float*)d_in[8];
  const float* mlp_w1    = (const float*)d_in[9];
  const float* mlp_b1    = (const float*)d_in[10];
  const float* mlp_w2    = (const float*)d_in[11];
  const float* mlp_b2    = (const float*)d_in[12];
  float* out = (float*)d_out;

  char* ws = (char*)d_ws;
  size_t off = 0;
  auto alloc = [&](size_t bytes) -> char* {
    char* p = ws + off;
    off += (bytes + 255) & ~(size_t)255;
    return p;
  };
  float*          tok  = (float*)alloc((size_t)Mrows * Cc * 4);
  unsigned short* xln  = (unsigned short*)alloc((size_t)Mrows * Cc * 2);
  unsigned short* qkvb = (unsigned short*)alloc((size_t)Mrows * 3 * Cc * 2);
  unsigned short* attn = (unsigned short*)alloc((size_t)Mrows * Cc * 2);
  unsigned short* hid  = (unsigned short*)alloc((size_t)Mrows * HIDc * 2);
  unsigned short* wqkv = (unsigned short*)alloc((size_t)3 * Cc * Cc * 2);
  unsigned short* wout = (unsigned short*)alloc((size_t)Cc * Cc * 2);
  unsigned short* w1   = (unsigned short*)alloc((size_t)HIDc * Cc * 2);
  unsigned short* w2   = (unsigned short*)alloc((size_t)Cc * HIDc * 2);

  // weight downcasts
  k_f32_to_bf16<<<(3 * Cc * Cc + 255) / 256, 256, 0, stream>>>(in_proj_w, wqkv, 3 * Cc * Cc);
  k_f32_to_bf16<<<(Cc * Cc + 255) / 256, 256, 0, stream>>>(out_w, wout, Cc * Cc);
  k_f32_to_bf16<<<(HIDc * Cc + 255) / 256, 256, 0, stream>>>(mlp_w1, w1, HIDc * Cc);
  k_f32_to_bf16<<<(Cc * HIDc + 255) / 256, 256, 0, stream>>>(mlp_w2, w2, Cc * HIDc);

  // LN1 (+ token transpose)
  k_ln1<<<Mrows, 256, 0, stream>>>(x, ln1_w, ln1_b, tok, xln);

  // QKV GEMM: [8192,512] x [1536,512]^T -> bf16 qkv
  k_gemm_bf16<<<dim3(3 * Cc / BN, Mrows / BM), 128, 0, stream>>>(
      xln, wqkv, in_proj_b, qkvb, nullptr, Mrows, 3 * Cc, Cc, 0);

  // attention: one block per (b, head, query time block)
  k_attn<<<Bc * NHc * Tt, 128, 0, stream>>>(qkvb, attn);

  // out-proj GEMM + residual into tok (fp32)
  k_gemm_bf16<<<dim3(Cc / BN, Mrows / BM), 128, 0, stream>>>(
      attn, wout, out_b, nullptr, tok, Mrows, Cc, Cc, 2);

  // LN2
  k_ln2<<<Mrows, 256, 0, stream>>>(tok, ln2_w, ln2_b, xln);

  // MLP1 + SiLU -> bf16 hidden
  k_gemm_bf16<<<dim3(HIDc / BN, Mrows / BM), 128, 0, stream>>>(
      xln, w1, mlp_b1, hid, nullptr, Mrows, HIDc, Cc, 1);

  // MLP2 + residual into tok
  k_gemm_bf16<<<dim3(Cc / BN, Mrows / BM), 128, 0, stream>>>(
      hid, w2, mlp_b2, nullptr, tok, Mrows, Cc, HIDc, 2);

  // transpose back to [B,C,T,H,W]
  k_out_transpose<<<((size_t)Bc * Cc * Sc + 255) / 256, 256, 0, stream>>>(tok, out);
}
